// MambaMixer_29248727286306
// MI455X (gfx1250) — compile-verified, matrix operand order checked by reading, not run on hardware
//
#include <hip/hip_runtime.h>
#include <hip/hip_bf16.h>
#include <math.h>

// ---------------- problem constants ----------------
#define B_ 2
#define L_ 1024
#define H_ 2048
#define D_ 4096
#define N_ 16
#define R_ 128
#define KW_ 4
#define M_ (B_ * L_)          // 2048 rows for all GEMMs

typedef __bf16 bf16;
typedef __attribute__((ext_vector_type(16))) __bf16 v16bf;
typedef __attribute__((ext_vector_type(8)))  __bf16 v8bf;
typedef __attribute__((ext_vector_type(8)))  float  v8f;

union V16U { v16bf v; v8bf h[2]; };

// ---------------- CDNA5 async global->LDS copy helpers ----------------
__device__ __forceinline__ void async_load_b128(unsigned lds_off, const bf16* gptr) {
    // VDST = LDS byte address (low 32 bits of flat shared pointer), VADDR = 64-bit global
    asm volatile("global_load_async_to_lds_b128 %0, %1, off"
                 :: "v"(lds_off), "v"(gptr) : "memory");
}
__device__ __forceinline__ void wait_async(int pending) {
    if (pending) asm volatile("s_wait_asynccnt 0x4" ::: "memory");
    else         asm volatile("s_wait_asynccnt 0x0" ::: "memory");
}

// ---------------- f32 -> bf16 convert ----------------
__global__ __launch_bounds__(256) void k_f32_to_bf16(const float* __restrict__ in,
                                                     bf16* __restrict__ out, int n) {
    int i = blockIdx.x * 256 + threadIdx.x;
    if (i < n) out[i] = (bf16)in[i];
}

// =====================================================================
// Macro-tiled bf16 WMMA GEMM: C[M,N] = A[M,K] @ W[N,K]^T
// 128x128 block tile, BK=32, double-buffered async LDS staging.
// Requires: M%128==0, N%128==0, K%32==0.
// ACT: 0 = none, 1 = softplus(x + bias[col])
// =====================================================================
template <int ACT>
__global__ __launch_bounds__(256) void k_gemm_bf16_mt(const bf16* __restrict__ A,
                                                      const bf16* __restrict__ W,
                                                      const float* __restrict__ bias,
                                                      float* __restrict__ C,
                                                      int M, int N, int K) {
    __shared__ __align__(16) bf16 sA[2][128 * 32];   // 16 KB
    __shared__ __align__(16) bf16 sB[2][128 * 32];   // 16 KB

    const int t    = threadIdx.x;
    const int wave = t >> 5;
    const int lane = t & 31;
    const int r    = lane & 15;      // row (A) / col (B) within 16x16 tile
    const int half = lane >> 4;      // K sub-block select per ISA bf16 layout
    const int wm   = wave >> 2;      // 0..1 : 64-row slab
    const int wn   = wave & 3;       // 0..3 : 32-col slab

    const int gm0 = blockIdx.y * 128;
    const int gn0 = blockIdx.x * 128;

    // ---- per-thread async copy slots: 512 16B-chunks per 128x32 panel ----
    const int cA0 = t, cA1 = t + 256;                 // chunk = row*4 + ch
    const bf16* gA0 = A + (size_t)(gm0 + (cA0 >> 2)) * K + (cA0 & 3) * 8;
    const bf16* gA1 = A + (size_t)(gm0 + (cA1 >> 2)) * K + (cA1 & 3) * 8;
    const bf16* gB0 = W + (size_t)(gn0 + (cA0 >> 2)) * K + (cA0 & 3) * 8;
    const bf16* gB1 = W + (size_t)(gn0 + (cA1 >> 2)) * K + (cA1 & 3) * 8;
    const unsigned sAoff = (unsigned)(uintptr_t)(&sA[0][0]);
    const unsigned sBoff = (unsigned)(uintptr_t)(&sB[0][0]);
    const unsigned lA0 = sAoff + (unsigned)(((cA0 >> 2) * 32 + (cA0 & 3) * 8) * 2);
    const unsigned lA1 = sAoff + (unsigned)(((cA1 >> 2) * 32 + (cA1 & 3) * 8) * 2);
    const unsigned lB0 = sBoff + (unsigned)(((cA0 >> 2) * 32 + (cA0 & 3) * 8) * 2);
    const unsigned lB1 = sBoff + (unsigned)(((cA1 >> 2) * 32 + (cA1 & 3) * 8) * 2);

    auto issue = [&](int buf, int k0) {
        const unsigned bofs = (unsigned)buf * (128 * 32 * 2);
        async_load_b128(lA0 + bofs, gA0 + k0);
        async_load_b128(lA1 + bofs, gA1 + k0);
        async_load_b128(lB0 + bofs, gB0 + k0);
        async_load_b128(lB1 + bofs, gB1 + k0);
    };

    v8f acc[4][2] = {};
    const int nsteps = K >> 5;

    issue(0, 0);
    for (int ks = 0; ks < nsteps; ++ks) {
        const int cur = ks & 1;
        const int more = (ks + 1 < nsteps);
        if (more) issue(cur ^ 1, (ks + 1) << 5);
        wait_async(more);
        __syncthreads();

        // ---- fragment loads from LDS (ds_load_b128 pairs, ISA layouts) ----
        V16U af[4], bf[2];
        const bf16* aBase = &sA[cur][0];
        const bf16* bBase = &sB[cur][0];
#pragma unroll
        for (int tr = 0; tr < 4; ++tr) {
            const int row = wm * 64 + tr * 16 + r;
            af[tr].h[0] = *(const v8bf*)(aBase + row * 32 + half * 8);
            af[tr].h[1] = *(const v8bf*)(aBase + row * 32 + 16 + half * 8);
        }
#pragma unroll
        for (int tc = 0; tc < 2; ++tc) {
            const int col = wn * 32 + tc * 16 + r;
            bf[tc].h[0] = *(const v8bf*)(bBase + col * 32 + half * 8);
            bf[tc].h[1] = *(const v8bf*)(bBase + col * 32 + 16 + half * 8);
        }
#pragma unroll
        for (int tr = 0; tr < 4; ++tr)
#pragma unroll
            for (int tc = 0; tc < 2; ++tc)
                acc[tr][tc] = __builtin_amdgcn_wmma_f32_16x16x32_bf16(
                    false, af[tr].v, false, bf[tc].v,
                    (short)0, acc[tr][tc], false, false);

        __syncthreads();
    }

    // ---- epilogue: C/D layout lanes0-15 -> N=lane,M=v; lanes16-31 -> N=lane-16,M=8+v
#pragma unroll
    for (int tr = 0; tr < 4; ++tr) {
#pragma unroll
        for (int tc = 0; tc < 2; ++tc) {
            const int col  = gn0 + wn * 32 + tc * 16 + r;
            const int row0 = gm0 + wm * 64 + tr * 16 + half * 8;
            float bv = (ACT == 1) ? bias[col] : 0.0f;
#pragma unroll
            for (int v = 0; v < 8; ++v) {
                float x = acc[tr][tc][v];
                if (ACT == 1) {
                    x += bv;
                    x = (x > 20.0f) ? x : log1pf(__expf(x));   // softplus
                }
                C[(size_t)(row0 + v) * N + col] = x;
            }
        }
    }
}

// ---------------- simple per-wave WMMA GEMM (for N=160 case) ----------------
template <int ACT>
__global__ __launch_bounds__(256) void k_gemm_bf16(const bf16* __restrict__ A,
                                                   const bf16* __restrict__ W,
                                                   const float* __restrict__ bias,
                                                   float* __restrict__ C,
                                                   int M, int N, int K) {
    const int wave = threadIdx.x >> 5;
    const int lane = threadIdx.x & 31;
    const int r    = lane & 15;
    const int half = lane >> 4;

    const int tilesN = N >> 4;
    const int total  = (M >> 4) * tilesN;
    const int gw     = blockIdx.x * 8 + wave;
    if (gw >= total) return;

    const int tM = gw / tilesN;
    const int tN = gw % tilesN;

    const bf16* aRow = A + (size_t)(tM * 16 + r) * K;
    const bf16* bRow = W + (size_t)(tN * 16 + r) * K;

    v8f acc = {};
    for (int k0 = 0; k0 < K; k0 += 32) {
        V16U a, b;
        a.h[0] = *(const v8bf*)(aRow + k0 + half * 8);
        a.h[1] = *(const v8bf*)(aRow + k0 + 16 + half * 8);
        b.h[0] = *(const v8bf*)(bRow + k0 + half * 8);
        b.h[1] = *(const v8bf*)(bRow + k0 + 16 + half * 8);
        acc = __builtin_amdgcn_wmma_f32_16x16x32_bf16(
            false, a.v, false, b.v, (short)0, acc, false, false);
    }

    const int col  = tN * 16 + r;
    const int row0 = tM * 16 + half * 8;
    float bv = (ACT == 1) ? bias[col] : 0.0f;
#pragma unroll
    for (int v = 0; v < 8; ++v) {
        float x = acc[v];
        if (ACT == 1) {
            x += bv;
            x = (x > 20.0f) ? x : log1pf(__expf(x));
        }
        C[(size_t)(row0 + v) * N + col] = x;
    }
}

// ---------------- causal depthwise conv (K=4) + SiLU, emits f32 + bf16 ----------------
__global__ __launch_bounds__(256) void k_conv_silu(const float* __restrict__ proj,
                                                   const float* __restrict__ cw,
                                                   const float* __restrict__ cb,
                                                   float* __restrict__ u,
                                                   bf16* __restrict__ ub) {
    int idx = blockIdx.x * 256 + threadIdx.x;        // over M_*D_
    if (idx >= M_ * D_) return;
    int d  = idx & (D_ - 1);
    int ml = idx >> 12;                               // D_ = 4096 = 2^12
    int l  = ml & (L_ - 1);
    int b  = ml >> 10;                                // L_ = 1024 = 2^10

    float acc = cb[d];
#pragma unroll
    for (int k = 0; k < KW_; ++k) {
        int lp = l - (KW_ - 1) + k;
        if (lp >= 0) {
            float hv = proj[((size_t)(b * L_ + lp)) * (2 * D_) + d];
            acc += cw[d * KW_ + k] * hv;
        }
    }
    float s = acc / (1.0f + __expf(-acc));            // SiLU
    u[idx]  = s;
    ub[idx] = (bf16)s;
}

// ---------------- extract dt columns (first R_ of ssm row) to bf16 ----------------
__global__ __launch_bounds__(256) void k_extract_dt(const float* __restrict__ ssm,
                                                    bf16* __restrict__ dtb) {
    int i = blockIdx.x * 256 + threadIdx.x;          // over M_*R_
    if (i >= M_ * R_) return;
    int m = i >> 7;                                   // R_ = 128
    int r = i & (R_ - 1);
    dtb[i] = (bf16)ssm[(size_t)m * (R_ + 2 * N_) + r];
}

// ---------------- sequential selective-scan over L ----------------
__global__ __launch_bounds__(256) void k_scan(const float* __restrict__ dt,
                                              const float* __restrict__ u,
                                              const float* __restrict__ ssm,
                                              const float* __restrict__ A_log,
                                              float* __restrict__ y) {
    __shared__ float sB[N_];
    __shared__ float sC[N_];

    const int tid  = threadIdx.x;
    const int b    = blockIdx.x >> 4;                 // 16 blocks per batch
    const int dblk = blockIdx.x & 15;
    const int d    = dblk * 256 + tid;

    float a[N_], s[N_];
#pragma unroll
    for (int n = 0; n < N_; ++n) {
        a[n] = -__expf(A_log[(size_t)d * N_ + n]);    // A = -exp(A_log)
        s[n] = 0.0f;
    }

    for (int l = 0; l < L_; ++l) {
        const size_t row = (size_t)b * L_ + l;
        if (tid < 32) {
            float v = ssm[row * (R_ + 2 * N_) + R_ + tid];
            if (tid < N_) sB[tid] = v; else sC[tid - N_] = v;
        }
        __syncthreads();

        float dtv = dt[row * D_ + d];
        float uv  = u[row * D_ + d];
        float dbu = dtv * uv;
        float acc = 0.0f;
#pragma unroll
        for (int n = 0; n < N_; ++n) {
            s[n] = s[n] * __expf(dtv * a[n]) + dbu * sB[n];
            acc += s[n] * sC[n];
        }
        y[row * D_ + d] = acc;
        __syncthreads();
    }
}

// ---------------- gating epilogue: y2 = (y + u*Dp) * silu(gate), to bf16 ----------------
__global__ __launch_bounds__(256) void k_gate(const float* __restrict__ y,
                                              const float* __restrict__ u,
                                              const float* __restrict__ proj,
                                              const float* __restrict__ Dp,
                                              bf16* __restrict__ y2b) {
    int idx = blockIdx.x * 256 + threadIdx.x;        // over M_*D_
    if (idx >= M_ * D_) return;
    int d = idx & (D_ - 1);
    int m = idx >> 12;
    float g  = proj[(size_t)m * (2 * D_) + D_ + d];
    float sg = g / (1.0f + __expf(-g));
    float v  = (y[idx] + u[idx] * Dp[d]) * sg;
    y2b[idx] = (bf16)v;
}

// ---------------- host side ----------------
extern "C" void kernel_launch(void* const* d_in, const int* in_sizes, int n_in,
                              void* d_out, int out_size, void* d_ws, size_t ws_size,
                              hipStream_t stream) {
    const float* x         = (const float*)d_in[0];  // [B,L,H]
    const float* in_proj_w = (const float*)d_in[1];  // [2D,H]
    const float* conv_w    = (const float*)d_in[2];  // [D,1,K]
    const float* conv_b    = (const float*)d_in[3];  // [D]
    const float* x_proj_w  = (const float*)d_in[4];  // [R+2N, D]
    const float* dt_proj_w = (const float*)d_in[5];  // [D, R]
    const float* dt_proj_b = (const float*)d_in[6];  // [D]
    const float* A_log     = (const float*)d_in[7];  // [D,N]
    const float* Dp        = (const float*)d_in[8];  // [D]
    const float* out_proj_w= (const float*)d_in[9];  // [H, D]
    float* out = (float*)d_out;                       // [B,L,H]

    char* ws = (char*)d_ws;
    size_t off = 0;
    auto alloc = [&](size_t bytes) -> char* {
        char* p = ws + off;
        off = (off + bytes + 255) & ~(size_t)255;
        return p;
    };

    float* proj  = (float*)alloc((size_t)M_ * 2 * D_ * 4);
    float* u     = (float*)alloc((size_t)M_ * D_ * 4);
    bf16*  ub    = (bf16*) alloc((size_t)M_ * D_ * 2);
    float* ssm   = (float*)alloc((size_t)M_ * (R_ + 2 * N_) * 4);
    bf16*  dtb   = (bf16*) alloc((size_t)M_ * R_ * 2);
    float* dt    = (float*)alloc((size_t)M_ * D_ * 4);
    float* y     = (float*)alloc((size_t)M_ * D_ * 4);
    bf16*  y2b   = (bf16*) alloc((size_t)M_ * D_ * 2);
    bf16*  xb    = (bf16*) alloc((size_t)M_ * H_ * 2);
    bf16*  w1b   = (bf16*) alloc((size_t)2 * D_ * H_ * 2);
    bf16*  xpwb  = (bf16*) alloc((size_t)(R_ + 2 * N_) * D_ * 2);
    bf16*  dtpwb = (bf16*) alloc((size_t)D_ * R_ * 2);
    bf16*  opwb  = (bf16*) alloc((size_t)H_ * D_ * 2);
    (void)ws_size; (void)in_sizes; (void)n_in; (void)out_size;

    auto cvt = [&](const float* src, bf16* dst, int n) {
        k_f32_to_bf16<<<(n + 255) / 256, 256, 0, stream>>>(src, dst, n);
    };
    cvt(x,          xb,    M_ * H_);
    cvt(in_proj_w,  w1b,   2 * D_ * H_);
    cvt(x_proj_w,   xpwb,  (R_ + 2 * N_) * D_);
    cvt(dt_proj_w,  dtpwb, D_ * R_);
    cvt(out_proj_w, opwb,  H_ * D_);

    // 1) in_proj: proj = x @ in_proj_w^T   [2048 x 8192, K=2048]
    k_gemm_bf16_mt<0><<<dim3((2 * D_) / 128, M_ / 128), 256, 0, stream>>>(
        xb, w1b, nullptr, proj, M_, 2 * D_, H_);

    // 2) causal depthwise conv + SiLU -> u (f32 + bf16)
    k_conv_silu<<<(M_ * D_ + 255) / 256, 256, 0, stream>>>(proj, conv_w, conv_b, u, ub);

    // 3) x_proj: ssm = u @ x_proj_w^T   [2048 x 160, K=4096]  (N=160: simple kernel)
    k_gemm_bf16<0><<<((M_ / 16) * ((R_ + 2 * N_) / 16) + 7) / 8, 256, 0, stream>>>(
        ub, xpwb, nullptr, ssm, M_, R_ + 2 * N_, D_);

    // 4) dt = softplus(ssm[:, :R] @ dt_proj_w^T + b)   [2048 x 4096, K=128]
    k_extract_dt<<<(M_ * R_ + 255) / 256, 256, 0, stream>>>(ssm, dtb);
    k_gemm_bf16_mt<1><<<dim3(D_ / 128, M_ / 128), 256, 0, stream>>>(
        dtb, dtpwb, dt_proj_b, dt, M_, D_, R_);

    // 5) sequential selective scan
    k_scan<<<B_ * (D_ / 256), 256, 0, stream>>>(dt, u, ssm, A_log, y);

    // 6) gating epilogue -> bf16
    k_gate<<<(M_ * D_ + 255) / 256, 256, 0, stream>>>(y, u, proj, Dp, y2b);

    // 7) out_proj: out = y2 @ out_proj_w^T   [2048 x 2048, K=4096]
    k_gemm_bf16_mt<0><<<dim3(H_ / 128, M_ / 128), 256, 0, stream>>>(
        y2b, opwb, nullptr, out, M_, H_, D_);
}
